// CausalGemmaMiniBitterLLM_49744311222842
// MI455X (gfx1250) — matrix operand order repaired and checked by reading, not compiled, not run on hardware
//
#include <hip/hip_runtime.h>

// ---------------- problem constants ----------------
#define BATCH 4
#define SEQ   2048
#define EMBED 1024
#define VOCAB 32000
#define TOKENS (BATCH * SEQ)          // 8192

// ---------------- d_out layout (floats, concatenated in return order) ------
#define OFF_LOGP   0LL                                    // B*S*V
#define OFF_PROBS  (OFF_LOGP   + (long long)TOKENS * VOCAB)
#define OFF_LOGITS (OFF_PROBS  + TOKENS)
#define OFF_GOUT   (OFF_LOGITS + TOKENS)
#define OFF_DSTO   (OFF_GOUT   + TOKENS)
#define OFF_UPDSTO (OFF_DSTO   + TOKENS)
#define OFF_NDSTO  (OFF_UPDSTO + TOKENS)
#define OFF_POSD   (OFF_NDSTO  + BATCH)

// ---------------- workspace layout (bytes) ----------------
#define WS_G        0x0000ULL                 // int[TOKENS]
#define WS_DST      0x8000ULL                 // int[TOKENS]
#define WS_UPDST    0x10000ULL                // int[TOKENS]
#define WS_SEGSTART 0x18000ULL                // int[TOKENS]
#define WS_NDST     0x20000ULL                // int[BATCH]
#define WS_XDOWN    0x40000ULL                // float[TOKENS*EMBED]   33.55 MB
#define WS_Y        (WS_XDOWN + (unsigned long long)TOKENS * EMBED * 4)   // bf16 y, 16.8 MB
#define WS_BW       (WS_Y + (unsigned long long)TOKENS * EMBED * 2)       // bf16 out_w^T, 65.5 MB

typedef __attribute__((ext_vector_type(16))) __bf16 v16bf;
typedef __attribute__((ext_vector_type(8)))  __bf16 v8bf;
typedef __attribute__((ext_vector_type(8)))  float  v8f;

// =====================================================================
// K1: gate logits / probs / g   (one wave of 32 lanes per token)
// =====================================================================
__global__ __launch_bounds__(256) void k_gate(const int* __restrict__ ids,
                                              const float* __restrict__ noise,
                                              const float* __restrict__ emb,
                                              const float* __restrict__ gw,
                                              const float* __restrict__ gb,
                                              int* __restrict__ ws_g,
                                              float* __restrict__ out) {
  int lane = threadIdx.x & 31;
  int wave = threadIdx.x >> 5;
  int t = blockIdx.x * 8 + wave;                // token id, 1024 blocks * 8 waves
  if (t >= TOKENS) return;
  long long id = ids[t];
  const float* row = emb + id * EMBED;
  float acc = 0.f;
  for (int d = lane; d < EMBED; d += 32) acc += row[d] * gw[d];
  for (int m = 16; m > 0; m >>= 1) acc += __shfl_xor(acc, m);
  if (lane == 0) {
    float logit = acc + gb[0];
    float prob  = 1.f / (1.f + __expf(-logit));
    int g = (noise[t] < prob) ? 1 : 0;
    if ((t % SEQ) == 0) g = 1;
    ws_g[t] = g;
    out[OFF_LOGITS + t] = logit;
    out[OFF_PROBS  + t] = prob;
    out[OFF_GOUT   + t] = (float)g;
  }
}

// =====================================================================
// K2: per-row scan -> down/up merge dst, n_dst, segment starts
// grid = BATCH blocks, 1024 threads (each owns 2 elements)
// =====================================================================
__global__ __launch_bounds__(1024) void k_scan(const int* __restrict__ ws_g,
                                               int* __restrict__ ws_dst,
                                               int* __restrict__ ws_updst,
                                               int* __restrict__ ws_segstart,
                                               int* __restrict__ ws_ndst,
                                               float* __restrict__ out) {
  __shared__ int garr[SEQ];
  __shared__ int ps[2][1024];
  __shared__ int carr[SEQ];
  int b = blockIdx.x, t = threadIdx.x;
  garr[t]        = ws_g[b * SEQ + t];
  garr[t + 1024] = ws_g[b * SEQ + t + 1024];
  __syncthreads();
  int g0 = garr[2 * t], g1 = garr[2 * t + 1];
  ps[0][t] = g0 + g1;
  __syncthreads();
  int src = 0;
  for (int off = 1; off < 1024; off <<= 1) {
    ps[1 - src][t] = ps[src][t] + (t >= off ? ps[src][t - off] : 0);
    __syncthreads();
    src = 1 - src;
  }
  int excl = (t > 0) ? ps[src][t - 1] : 0;
  carr[2 * t]     = excl + g0;
  carr[2 * t + 1] = excl + g0 + g1;
  __syncthreads();
  for (int r = 0; r < 2; ++r) {
    int i = 2 * t + r;
    int gi = garr[i];
    int dsti = carr[i] - gi;                       // exclusive cumsum
    ws_dst[b * SEQ + i] = dsti;
    out[OFF_DSTO + b * SEQ + i] = (float)dsti;
    int updsti = (i < SEQ - 1) ? (carr[i + 1] - 1 - garr[i + 1])
                               : (carr[SEQ - 1] - 1);
    ws_updst[b * SEQ + i] = updsti;
    out[OFF_UPDSTO + b * SEQ + i] = (float)updsti;
    if (i == 0 || garr[i - 1] == 1) ws_segstart[b * SEQ + dsti] = i;
    if (i == SEQ - 1) {
      ws_ndst[b] = dsti + 1;
      out[OFF_NDSTO + b] = (float)(dsti + 1);
    }
  }
}

// =====================================================================
// K3: segment means (x_down, pos_down). One wave per (b, segment j).
// Segments are contiguous token runs, so pos mean = (st+en-1)/2.
// =====================================================================
__global__ __launch_bounds__(256) void k_segmean(const int* __restrict__ ids,
                                                 const float* __restrict__ emb,
                                                 const int* __restrict__ ws_segstart,
                                                 const int* __restrict__ ws_ndst,
                                                 float* __restrict__ xdown,
                                                 float* __restrict__ out) {
  int lane = threadIdx.x & 31;
  int wave = threadIdx.x >> 5;
  int gid = blockIdx.x * 8 + wave;            // 1024 blocks * 8 waves = 8192
  int b = gid >> 11;                          // / SEQ
  int j = gid & (SEQ - 1);
  int n = ws_ndst[b];
  if (j >= n) {
    if (lane == 0) out[OFF_POSD + b * SEQ + j] = 0.f;   // empty segment -> 0
    return;
  }
  int st = ws_segstart[b * SEQ + j];
  int en = (j + 1 < n) ? ws_segstart[b * SEQ + j + 1] : SEQ;
  float inv = 1.f / (float)(en - st);
  float acc[32];
  #pragma unroll
  for (int c = 0; c < 32; ++c) acc[c] = 0.f;
  for (int s = st; s < en; ++s) {
    long long id = ids[b * SEQ + s];
    const float* row = emb + id * EMBED;
    #pragma unroll
    for (int c = 0; c < 32; ++c) acc[c] += row[c * 32 + lane];
  }
  float* xd = xdown + (long long)gid * EMBED;
  #pragma unroll
  for (int c = 0; c < 32; ++c) xd[c * 32 + lane] = acc[c] * inv;
  if (lane == 0) out[OFF_POSD + b * SEQ + j] = 0.5f * (float)(st + en - 1);
}

// =====================================================================
// K4: y = x + x_down[up_dst]  -> bf16 (GEMM A operand). One block/token.
// =====================================================================
__global__ __launch_bounds__(256) void k_makey(const int* __restrict__ ids,
                                               const int* __restrict__ ws_updst,
                                               const float* __restrict__ emb,
                                               const float* __restrict__ xdown,
                                               __bf16* __restrict__ Y) {
  int t = blockIdx.x;
  int b = t >> 11;
  long long id = ids[t];
  int ud = ws_updst[t];
  const float* xr = emb + id * EMBED;
  const float* dr = xdown + ((long long)b * SEQ + ud) * EMBED;
  __bf16* yr = Y + (long long)t * EMBED;
  for (int d = threadIdx.x; d < EMBED; d += 256)
    yr[d] = (__bf16)(xr[d] + dr[d]);
}

// =====================================================================
// K5: out_w (K-major fp32) -> Bw (N-major bf16) via LDS tile transpose
// grid = (VOCAB/32, EMBED/32)
// =====================================================================
__global__ __launch_bounds__(256) void k_transw(const float* __restrict__ ow,
                                                __bf16* __restrict__ Bw) {
  __shared__ float tile[32][33];
  int k0 = blockIdx.y * 32, n0 = blockIdx.x * 32;
  for (int r = 0; r < 4; ++r) {
    int idx = threadIdx.x + r * 256;
    int kr = idx >> 5, nc = idx & 31;
    tile[kr][nc] = ow[(long long)(k0 + kr) * VOCAB + n0 + nc];
  }
  __syncthreads();
  for (int r = 0; r < 4; ++r) {
    int idx = threadIdx.x + r * 256;
    int nr = idx >> 5, kc = idx & 31;
    Bw[(long long)(n0 + nr) * EMBED + k0 + kc] = (__bf16)tile[kc][nr];
  }
}

// =====================================================================
// K6: GEMM logits = Y(8192x1024) * Bw^T(n-major) + out_b
// 128x128 C tile, BK=32, 8 waves (2x4), async global->LDS double buffer,
// v_wmma_f32_16x16x32_bf16 compute.
// =====================================================================
#define BM 128
#define BN 128
#define BK 32

__global__ __launch_bounds__(256) void k_gemm(const __bf16* __restrict__ Y,
                                              const __bf16* __restrict__ Bw,
                                              const float* __restrict__ ob,
                                              float* __restrict__ logits) {
  __shared__ __bf16 sA[2][BM][BK];
  __shared__ __bf16 sB[2][BN][BK];
  const int tid = threadIdx.x;
  const int lane = tid & 31, wave = tid >> 5;
  const int wm = wave >> 2;          // 0..1 -> 64 rows each
  const int wn = wave & 3;           // 0..3 -> 32 cols each
  const int l16 = lane & 15, lhi = lane >> 4;
  const long long blockM = (long long)blockIdx.y * BM;
  const long long blockN = (long long)blockIdx.x * BN;

  v8f acc[4][2];
  v8f zero = {0.f, 0.f, 0.f, 0.f, 0.f, 0.f, 0.f, 0.f};
  #pragma unroll
  for (int mf = 0; mf < 4; ++mf)
    #pragma unroll
    for (int nf = 0; nf < 2; ++nf) acc[mf][nf] = zero;

  auto stage = [&](int kt, int buf) {
    const long long k0 = (long long)kt * BK;
    #pragma unroll
    for (int r = 0; r < 2; ++r) {                 // A tile: 128 rows x 4 16B chunks
      int idx = tid + r * 256;
      int row = idx >> 2, ck = idx & 3;
      unsigned lds = (unsigned)(unsigned long long)(void*)&sA[buf][row][ck * 8];
      unsigned long long ga =
          (unsigned long long)(const void*)(Y + (blockM + row) * EMBED + k0 + ck * 8);
      asm volatile("global_load_async_to_lds_b128 %0, %1, off"
                   :: "v"(lds), "v"(ga) : "memory");
    }
    #pragma unroll
    for (int r = 0; r < 2; ++r) {                 // B tile: 128 cols x 4 16B chunks
      int idx = tid + r * 256;
      int row = idx >> 2, ck = idx & 3;
      unsigned lds = (unsigned)(unsigned long long)(void*)&sB[buf][row][ck * 8];
      unsigned long long ga =
          (unsigned long long)(const void*)(Bw + (blockN + row) * EMBED + k0 + ck * 8);
      asm volatile("global_load_async_to_lds_b128 %0, %1, off"
                   :: "v"(lds), "v"(ga) : "memory");
    }
  };

  stage(0, 0);
  const int KT = EMBED / BK;                       // 32
  for (int kt = 0; kt < KT; ++kt) {
    int cur = kt & 1;
    asm volatile("s_wait_asynccnt 0" ::: "memory");
    __syncthreads();
    if (kt + 1 < KT) stage(kt + 1, cur ^ 1);

    // A fragments: lane l -> row l&15, K chunks [8*lhi, +8) and [16+8*lhi, +8)
    v16bf afrag[4];
    #pragma unroll
    for (int mf = 0; mf < 4; ++mf) {
      int m = wm * 64 + mf * 16 + l16;
      v8bf lo = *(const v8bf*)&sA[cur][m][lhi * 8];
      v8bf hi = *(const v8bf*)&sA[cur][m][16 + lhi * 8];
      afrag[mf] = __builtin_shufflevector(lo, hi, 0, 1, 2, 3, 4, 5, 6, 7,
                                          8, 9, 10, 11, 12, 13, 14, 15);
    }
    // B fragments: lane l -> col l&15, K chunks [8*lhi, +8) and [16+8*lhi, +8)
    v16bf bfrag[2];
    #pragma unroll
    for (int nf = 0; nf < 2; ++nf) {
      int n = wn * 32 + nf * 16 + l16;
      v8bf lo = *(const v8bf*)&sB[cur][n][lhi * 8];
      v8bf hi = *(const v8bf*)&sB[cur][n][16 + lhi * 8];
      bfrag[nf] = __builtin_shufflevector(lo, hi, 0, 1, 2, 3, 4, 5, 6, 7,
                                          8, 9, 10, 11, 12, 13, 14, 15);
    }
    #pragma unroll
    for (int mf = 0; mf < 4; ++mf)
      #pragma unroll
      for (int nf = 0; nf < 2; ++nf)
        acc[mf][nf] = __builtin_amdgcn_wmma_f32_16x16x32_bf16(
            false, afrag[mf], false, bfrag[nf], (short)0, acc[mf][nf],
            false, false);
    __syncthreads();
  }

  // epilogue: D layout -> lane l holds (M = r + 8*lhi, N = l&15) in vgpr r
  #pragma unroll
  for (int mf = 0; mf < 4; ++mf) {
    #pragma unroll
    for (int nf = 0; nf < 2; ++nf) {
      long long col = blockN + wn * 32 + nf * 16 + l16;
      float bias = ob[col];
      #pragma unroll
      for (int r = 0; r < 8; ++r) {
        long long row = blockM + wm * 64 + mf * 16 + lhi * 8 + r;
        logits[row * VOCAB + col] = acc[mf][nf][r] + bias;
      }
    }
  }
}

// =====================================================================
// K7: in-place online log_softmax, one block per row of 32000
// =====================================================================
__global__ __launch_bounds__(256) void k_logsoftmax(float* __restrict__ logp) {
  __shared__ float sm[256], ss[256];
  float* p = logp + (long long)blockIdx.x * VOCAB;
  int tid = threadIdx.x;
  float m = -3.4e38f, s = 0.f;
  for (int i = tid; i < VOCAB; i += 256) {
    float x = p[i];
    float nm = fmaxf(m, x);
    s = s * __expf(m - nm) + __expf(x - nm);
    m = nm;
  }
  sm[tid] = m; ss[tid] = s;
  __syncthreads();
  for (int off = 128; off > 0; off >>= 1) {
    if (tid < off) {
      float m2 = sm[tid + off], s2 = ss[tid + off];
      float nm = fmaxf(sm[tid], m2);
      ss[tid] = ss[tid] * __expf(sm[tid] - nm) + s2 * __expf(m2 - nm);
      sm[tid] = nm;
    }
    __syncthreads();
  }
  float M = sm[0];
  float L = logf(ss[0]);
  __syncthreads();
  for (int i = tid; i < VOCAB; i += 256) p[i] = p[i] - M - L;
}

// =====================================================================
extern "C" void kernel_launch(void* const* d_in, const int* in_sizes, int n_in,
                              void* d_out, int out_size, void* d_ws, size_t ws_size,
                              hipStream_t stream) {
  const int*   ids   = (const int*)d_in[0];
  const float* noise = (const float*)d_in[1];
  const float* emb   = (const float*)d_in[2];
  const float* gw    = (const float*)d_in[3];
  const float* gb    = (const float*)d_in[4];
  const float* ow    = (const float*)d_in[5];
  const float* ob    = (const float*)d_in[6];
  float* out = (float*)d_out;

  char* ws = (char*)d_ws;
  int*    ws_g        = (int*)(ws + WS_G);
  int*    ws_dst      = (int*)(ws + WS_DST);
  int*    ws_updst    = (int*)(ws + WS_UPDST);
  int*    ws_segstart = (int*)(ws + WS_SEGSTART);
  int*    ws_ndst     = (int*)(ws + WS_NDST);
  float*  xdown       = (float*)(ws + WS_XDOWN);
  __bf16* Y           = (__bf16*)(ws + WS_Y);
  __bf16* Bw          = (__bf16*)(ws + WS_BW);

  // weight conversion is independent of token path -> issue first
  k_transw<<<dim3(VOCAB / 32, EMBED / 32), 256, 0, stream>>>(ow, Bw);
  k_gate<<<TOKENS / 8, 256, 0, stream>>>(ids, noise, emb, gw, gb, ws_g, out);
  k_scan<<<BATCH, 1024, 0, stream>>>(ws_g, ws_dst, ws_updst, ws_segstart,
                                     ws_ndst, out);
  k_segmean<<<TOKENS / 8, 256, 0, stream>>>(ids, emb, ws_segstart, ws_ndst,
                                            xdown, out);
  k_makey<<<TOKENS, 256, 0, stream>>>(ids, ws_updst, emb, xdown, Y);
  k_gemm<<<dim3(VOCAB / BN, TOKENS / BM), 256, 0, stream>>>(Y, Bw, ob,
                                                            out + OFF_LOGP);
  k_logsoftmax<<<TOKENS, 256, 0, stream>>>(out + OFF_LOGP);
}
